// VectorQuantizerEMA_47949014892599
// MI455X (gfx1250) — matrix operand order repaired and checked by reading, not compiled, not run on hardware
//
#include <hip/hip_runtime.h>
#include <hip/hip_bf16.h>
#include <math.h>

typedef __attribute__((ext_vector_type(16))) __bf16 v16bf;
typedef __attribute__((ext_vector_type(8)))  __bf16 v8bf;
typedef __attribute__((ext_vector_type(8)))  float  v8f;

#define VQ_B 16
#define VQ_D 256
#define VQ_T 4096
#define VQ_K 1024
#define VQ_N (VQ_B * VQ_T)                    /* 65536 tokens */
#define VQ_ELEMS ((size_t)VQ_B * VQ_D * VQ_T) /* 16777216 */

#define LDS_STRIDE 264                        /* 256 + 8 pad, bf16 elems */
#define CODES_PER_STAGE 64
#define NSTAGES (VQ_K / CODES_PER_STAGE)      /* 16 */
#define STAGE_ELEMS (CODES_PER_STAGE * LDS_STRIDE)  /* 16896 bf16 per array */
#define BUF_ELEMS (2 * STAGE_ELEMS)                 /* hi + lo per buffer   */
#define VQ_LDS_BYTES (2 * BUF_ELEMS * 2)            /* 2 buffers = 135168 B */

// --- CDNA5 async global->LDS path (probe via __has_builtin, sync fallback) --
#if defined(__has_builtin)
#if __has_builtin(__builtin_amdgcn_global_load_async_to_lds_b128)
#define VQ_HAS_ASYNC 1
#endif
#endif
#ifndef VQ_HAS_ASYNC
#define VQ_HAS_ASYNC 0
#endif

#if VQ_HAS_ASYNC
// exact parameter types per hipcc diagnostic:
//   arg0: int __vector_size(16) __device__ *   (addrspace(1))
//   arg1: int __vector_size(16) __shared__ *   (addrspace(3))
typedef int vq_v4i __attribute__((vector_size(16)));
typedef vq_v4i __attribute__((address_space(1))) vq_g4i;
typedef vq_v4i __attribute__((address_space(3))) vq_l4i;
#endif

__device__ __forceinline__ void vq_wait_async() {
#if VQ_HAS_ASYNC
#if __has_builtin(__builtin_amdgcn_s_wait_asynccnt)
  __builtin_amdgcn_s_wait_asynccnt(0);
#else
  asm volatile("s_wait_asynccnt 0" ::: "memory");
#endif
#endif
}

// ---------------------------------------------------------------------------
// K0: zero the histogram counters
// ---------------------------------------------------------------------------
__global__ __launch_bounds__(256) void vq_init(unsigned* counts) {
  int i = blockIdx.x * 256 + threadIdx.x;
  if (i < VQ_K) counts[i] = 0u;
}

// ---------------------------------------------------------------------------
// K1: split embeddings into bf16 hi/lo (RNE split) and compute ||e_k||^2
// ---------------------------------------------------------------------------
__global__ __launch_bounds__(256) void vq_prep(const float* __restrict__ E,
                                               __bf16* __restrict__ ehi,
                                               __bf16* __restrict__ elo,
                                               float* __restrict__ en2) {
  const int k = blockIdx.x;
  const int d = threadIdx.x;
  float x = E[(size_t)k * VQ_D + d];
  __bf16 h = (__bf16)x;
  ehi[(size_t)k * VQ_D + d] = h;
  elo[(size_t)k * VQ_D + d] = (__bf16)(x - (float)h);

  float sq = x * x;
  #pragma unroll
  for (int m = 16; m >= 1; m >>= 1) sq += __shfl_xor(sq, m, 32);
  __shared__ float w[8];
  if ((d & 31) == 0) w[d >> 5] = sq;
  __syncthreads();
  if (d == 0) {
    float s = 0.f;
    #pragma unroll
    for (int i = 0; i < 8; ++i) s += w[i];
    en2[k] = s;
  }
}

// ---------------------------------------------------------------------------
// helpers for K2
// ---------------------------------------------------------------------------
__device__ __forceinline__ v16bf vq_ld16(const __bf16* p) {
  v8bf a = *(const v8bf*)p;
  v8bf b = *(const v8bf*)(p + 8);
  v16bf r;
  #pragma unroll
  for (int j = 0; j < 8; ++j) { r[j] = a[j]; r[j + 8] = b[j]; }
  return r;
}

// Stage 64 codes (hi+lo) into padded LDS. Async (ASYNCcnt, no VGPR round-trip)
// when available; plain vector copies otherwise.
__device__ __forceinline__ void vq_stage(const __bf16* __restrict__ ehi,
                                         const __bf16* __restrict__ elo,
                                         __bf16* shi, __bf16* slo,
                                         int kbase, int tid) {
  for (int i = tid; i < CODES_PER_STAGE * 32; i += 256) {  // 32 x16B per row
    const int code = i >> 5, c8 = i & 31;
    const size_t gof = (size_t)(kbase + code) * VQ_D + (size_t)c8 * 8;
    const int    lof = code * LDS_STRIDE + c8 * 8;
#if VQ_HAS_ASYNC
    __builtin_amdgcn_global_load_async_to_lds_b128(
        (vq_g4i*)(void*)(__bf16*)(ehi + gof),
        (vq_l4i*)(void*)(shi + lof), 0, 0);
    __builtin_amdgcn_global_load_async_to_lds_b128(
        (vq_g4i*)(void*)(__bf16*)(elo + gof),
        (vq_l4i*)(void*)(slo + lof), 0, 0);
#else
    *(v8bf*)(shi + lof) = *(const v8bf*)(ehi + gof);
    *(v8bf*)(slo + lof) = *(const v8bf*)(elo + gof);
#endif
  }
}

// ---------------------------------------------------------------------------
// K2: fused split-bf16 WMMA GEMM + argmin.
//     block = 256 threads = 8 waves; each wave owns 16 tokens (full D in regs,
//     hi+lo). Double-buffered 64-code LDS stages (async copies overlap the 96
//     WMMAs of the current stage). Two accumulator chains per pass for ILP.
// ---------------------------------------------------------------------------
__global__ __launch_bounds__(256) void vq_argmin(const float* __restrict__ z,
                                                 const __bf16* __restrict__ ehi,
                                                 const __bf16* __restrict__ elo,
                                                 const float* __restrict__ en2,
                                                 int* __restrict__ idx_out) {
  extern __shared__ char smem[];
  __bf16* bufs = (__bf16*)smem;   // [2][2][CODES_PER_STAGE][LDS_STRIDE]

  const int tid  = threadIdx.x;
  const int lane = tid & 31;
  const int wave = tid >> 5;
  const int half = lane >> 4;     // which 16-lane half
  const int l15  = lane & 15;

  // token (row) handled by this lane's half of the A layout
  const int row = blockIdx.x * 128 + wave * 16 + l15;
  const int bb  = row >> 12;      // row / 4096
  const int tt  = row & 4095;
  const size_t zbase = (size_t)bb * VQ_D * VQ_T + tt;  // z[(b*D+d)*T + t], d=0

  // ---- load A: 16 tokens x 256 dims, split bf16 hi/lo, contiguous k order (16*half + s) ----
  // lane<16 : elems 0..15 = K kt*32+0..15
  // lane>=16: elems 0..15 = K kt*32+16..31
  v16bf ahi[8], alo[8];
  #pragma unroll
  for (int kt = 0; kt < 8; ++kt) {
    const int k0 = kt * 32 + half * 16;
    const int k1 = kt * 32 + half * 16 + 8;
    #pragma unroll
    for (int j = 0; j < 8; ++j) {
      float x0 = z[zbase + (size_t)(k0 + j) * VQ_T];
      __bf16 h0 = (__bf16)x0;
      ahi[kt][j] = h0;
      alo[kt][j] = (__bf16)(x0 - (float)h0);
      float x1 = z[zbase + (size_t)(k1 + j) * VQ_T];
      __bf16 h1 = (__bf16)x1;
      ahi[kt][j + 8] = h1;
      alo[kt][j + 8] = (__bf16)(x1 - (float)h1);
    }
  }

  float minv[8];
  int   mini[8];
  #pragma unroll
  for (int r = 0; r < 8; ++r) { minv[r] = 3.4e38f; mini[r] = 0; }

  // prologue: stage 0 into buffer 0
  vq_stage(ehi, elo, bufs, bufs + STAGE_ELEMS, 0, tid);
  vq_wait_async();
  __syncthreads();

  for (int kb = 0; kb < NSTAGES; ++kb) {
    const int cur = kb & 1;
    __bf16* shi = bufs + cur * BUF_ELEMS;
    __bf16* slo = shi + STAGE_ELEMS;

    // kick off next stage into the other buffer (overlaps with compute below)
    if (kb + 1 < NSTAGES) {
      __bf16* nhi = bufs + (1 - cur) * BUF_ELEMS;
      vq_stage(ehi, elo, nhi, nhi + STAGE_ELEMS,
               (kb + 1) * CODES_PER_STAGE, tid);
    }

    const int kbase = kb * CODES_PER_STAGE;
    #pragma unroll
    for (int nc2 = 0; nc2 < CODES_PER_STAGE / 32; ++nc2) {
      const int colA  = nc2 * 32 + l15;       // two 16-col chunks -> two
      const int colB  = colA + 16;            // independent WMMA chains
      const float e2a = en2[kbase + colA];
      const float e2b = en2[kbase + colB];

      v8f c0 = {0.f, 0.f, 0.f, 0.f, 0.f, 0.f, 0.f, 0.f};
      v8f c1 = {0.f, 0.f, 0.f, 0.f, 0.f, 0.f, 0.f, 0.f};
      #pragma unroll
      for (int kt = 0; kt < 8; ++kt) {
        // B layout: lane<16 holds K kt*32+0..15 of its column; lane>=16 holds
        // K kt*32+16..31 of the same column.
        const int koff = kt * 32 + half * 16;
        v16bf bh0 = vq_ld16(shi + colA * LDS_STRIDE + koff);
        v16bf bl0 = vq_ld16(slo + colA * LDS_STRIDE + koff);
        v16bf bh1 = vq_ld16(shi + colB * LDS_STRIDE + koff);
        v16bf bl1 = vq_ld16(slo + colB * LDS_STRIDE + koff);
        // dot += zhi*ehi + zhi*elo + zlo*ehi  (split-bf16 ~ fp32 accuracy)
        c0 = __builtin_amdgcn_wmma_f32_16x16x32_bf16(false, ahi[kt], false, bh0,
                                                     (short)0, c0, false, false);
        c1 = __builtin_amdgcn_wmma_f32_16x16x32_bf16(false, ahi[kt], false, bh1,
                                                     (short)0, c1, false, false);
        c0 = __builtin_amdgcn_wmma_f32_16x16x32_bf16(false, ahi[kt], false, bl0,
                                                     (short)0, c0, false, false);
        c1 = __builtin_amdgcn_wmma_f32_16x16x32_bf16(false, ahi[kt], false, bl1,
                                                     (short)0, c1, false, false);
        c0 = __builtin_amdgcn_wmma_f32_16x16x32_bf16(false, alo[kt], false, bh0,
                                                     (short)0, c0, false, false);
        c1 = __builtin_amdgcn_wmma_f32_16x16x32_bf16(false, alo[kt], false, bh1,
                                                     (short)0, c1, false, false);
      }
      // distance = ||e||^2 - 2*dot ; ||z||^2 is per-row constant (irrelevant).
      // colA checked before colB keeps first-occurrence tie semantics.
      #pragma unroll
      for (int r = 0; r < 8; ++r) {
        float v0 = e2a - 2.0f * c0[r];
        if (v0 < minv[r]) { minv[r] = v0; mini[r] = kbase + colA; }
        float v1 = e2b - 2.0f * c1[r];
        if (v1 < minv[r]) { minv[r] = v1; mini[r] = kbase + colB; }
      }
    }

    vq_wait_async();   // next stage fully in LDS before anyone reads it
    __syncthreads();
  }

  // cross-lane argmin within each 16-lane half (C layout: lanes = columns)
  #pragma unroll
  for (int r = 0; r < 8; ++r) {
    float v = minv[r];
    int   ix = mini[r];
    #pragma unroll
    for (int m = 8; m >= 1; m >>= 1) {
      float ov = __shfl_xor(v, m, 32);
      int   oi = __shfl_xor(ix, m, 32);
      if (ov < v || (ov == v && oi < ix)) { v = ov; ix = oi; }
    }
    if (l15 == 0) {
      // VGPR r: lanes 0-15 -> row base+r ; lanes 16-31 -> row base+r+8
      const int orow = blockIdx.x * 128 + wave * 16 + r + half * 8;
      idx_out[orow] = ix;
    }
  }
}

// ---------------------------------------------------------------------------
// K3: gather quantized output in (B,D,T) layout + per-block sum of squared
//     error (deterministic two-stage reduction, no float atomics)
// ---------------------------------------------------------------------------
__global__ __launch_bounds__(256) void vq_gather(const float* __restrict__ z,
                                                 const float* __restrict__ E,
                                                 const int* __restrict__ idx,
                                                 float* __restrict__ outQ,
                                                 float* __restrict__ bsums) {
  const int lin = blockIdx.x * 256 + threadIdx.x;   // over B*D*T = 2^24
  const int t = lin & 4095;
  const int d = (lin >> 12) & 255;
  const int b = lin >> 20;
  const int n = (b << 12) + t;
  const int k = idx[n];
  const float val = E[(size_t)k * VQ_D + d];
  outQ[lin] = val;                                   // STE fwd == quantized
  float df = val - z[lin];
  float sq = df * df;
  #pragma unroll
  for (int m = 16; m >= 1; m >>= 1) sq += __shfl_xor(sq, m, 32);
  __shared__ float w[8];
  if ((threadIdx.x & 31) == 0) w[threadIdx.x >> 5] = sq;
  __syncthreads();
  if (threadIdx.x == 0) {
    float s = 0.f;
    #pragma unroll
    for (int i = 0; i < 8; ++i) s += w[i];
    bsums[blockIdx.x] = s;
  }
}

// ---------------------------------------------------------------------------
// K4: integer histogram (exact, deterministic) + emit indices as floats
// ---------------------------------------------------------------------------
__global__ __launch_bounds__(256) void vq_hist(const int* __restrict__ idx,
                                               unsigned* __restrict__ counts,
                                               float* __restrict__ outI) {
  const int n = blockIdx.x * 256 + threadIdx.x;
  const int k = idx[n];
  atomicAdd(&counts[k], 1u);
  outI[n] = (float)k;
}

// ---------------------------------------------------------------------------
// K5: final loss + perplexity (single block, fixed-order reductions)
// ---------------------------------------------------------------------------
__global__ __launch_bounds__(256) void vq_finalize(const float* __restrict__ bsums,
                                                   const unsigned* __restrict__ counts,
                                                   float* __restrict__ outScalars) {
  const int tid = threadIdx.x;
  __shared__ float red[256];

  float s = 0.f;
  for (int i = tid; i < (int)(VQ_ELEMS / 256); i += 256) s += bsums[i];
  red[tid] = s;
  __syncthreads();
  for (int m = 128; m >= 1; m >>= 1) {
    if (tid < m) red[tid] += red[tid + m];
    __syncthreads();
  }
  const float lossSum = red[0];
  __syncthreads();

  float h = 0.f;
  for (int i = tid; i < VQ_K; i += 256) {
    float p = (float)counts[i] * (1.0f / (float)VQ_N);
    h += p * logf(p + 1e-10f);
  }
  red[tid] = h;
  __syncthreads();
  for (int m = 128; m >= 1; m >>= 1) {
    if (tid < m) red[tid] += red[tid + m];
    __syncthreads();
  }
  if (tid == 0) {
    outScalars[0] = 0.25f * lossSum / (float)VQ_ELEMS;  // commitment loss
    outScalars[1] = expf(-red[0]);                      // perplexity
  }
}

// ---------------------------------------------------------------------------
extern "C" void kernel_launch(void* const* d_in, const int* in_sizes, int n_in,
                              void* d_out, int out_size, void* d_ws, size_t ws_size,
                              hipStream_t stream) {
  const float* z = (const float*)d_in[0];      // (B, D, T) fp32
  const float* E = (const float*)d_in[1];      // (K, D)    fp32
  float* out = (float*)d_out;

  // workspace layout (~1.6 MB)
  char* ws = (char*)d_ws;
  __bf16*   ehi    = (__bf16*)(ws + 0);          // 512 KB
  __bf16*   elo    = (__bf16*)(ws + 524288);     // 512 KB
  float*    en2    = (float*)(ws + 1048576);     // 4 KB
  unsigned* counts = (unsigned*)(ws + 1052672);  // 4 KB
  int*      idxw   = (int*)(ws + 1056768);       // 256 KB
  float*    bsums  = (float*)(ws + 1318912);     // 256 KB

  float* outQ = out;                       // B*D*T quantized (STE fwd)
  float* outI = out + VQ_ELEMS;            // B*T indices (as float)
  float* outS = outI + VQ_N;               // loss, perplexity

  vq_init<<<4, 256, 0, stream>>>(counts);
  vq_prep<<<VQ_K, 256, 0, stream>>>(E, ehi, elo, en2);
  vq_argmin<<<VQ_N / 128, 256, VQ_LDS_BYTES, stream>>>(z, ehi, elo, en2, idxw);
  vq_gather<<<(int)(VQ_ELEMS / 256), 256, 0, stream>>>(z, E, idxw, outQ, bsums);
  vq_hist<<<VQ_N / 256, 256, 0, stream>>>(idxw, counts, outI);
  vq_finalize<<<1, 256, 0, stream>>>(bsums, counts, outS);
}
